// GINModel_71227737636882
// MI455X (gfx1250) — matrix operand order repaired and checked by reading, not compiled, not run on hardware
//
#include <hip/hip_runtime.h>

// ---------------------------------------------------------------------------
// Types for CDNA5 WMMA / TDM
// ---------------------------------------------------------------------------
typedef __attribute__((ext_vector_type(16))) __bf16 v16bf;
typedef __attribute__((ext_vector_type(2)))  __bf16 v2bf;
typedef __attribute__((ext_vector_type(8)))  float  v8f;
typedef __attribute__((ext_vector_type(4)))  unsigned int v4u;
typedef __attribute__((ext_vector_type(8)))  int    v8i_t;
typedef __attribute__((ext_vector_type(4)))  int    v4i_t;

#if defined(__has_builtin)
#if __has_builtin(__builtin_amdgcn_tensor_load_to_lds)
#define HAVE_TDM 1
#endif
#if __has_builtin(__builtin_amdgcn_s_wait_tensorcnt)
#define HAVE_TENSORCNT 1
#endif
#endif

#if defined(__has_include)
#if __has_include(<hip/amd_detail/amd_gfx1250_TDM.h>)
#define TDM_6ARG 1   // therock-10.0 headers -> 6-arg builtin
#endif
#endif

// Native f32->bf16 (RNE fptrunc; backend picks v_cvt_*bf16* instructions)
__device__ inline unsigned short f32_to_bf16_bits(float f) {
    return __builtin_bit_cast(unsigned short, (__bf16)f);
}
__device__ inline unsigned pack_bf16x2(float lo, float hi) {
    v2bf p;
    p.x = (__bf16)lo;
    p.y = (__bf16)hi;
    return __builtin_bit_cast(unsigned, p);
}

// ---------------------------------------------------------------------------
// f32 [K][Nout] -> bf16 bits transposed [Nout][K]
// ---------------------------------------------------------------------------
__global__ void cvt_bf16_transpose_kernel(const float* __restrict__ src,
                                          unsigned short* __restrict__ dst,
                                          int K, int Nout) {
    const int n = K * Nout;
    for (int i = blockIdx.x * blockDim.x + threadIdx.x; i < n;
         i += gridDim.x * blockDim.x) {
        const int k  = i / Nout;
        const int nn = i - k * Nout;
        dst[(size_t)nn * K + k] = f32_to_bf16_bits(src[i]);
    }
}

// ---------------------------------------------------------------------------
// dst = src (float copy, grid-stride, float4)
// ---------------------------------------------------------------------------
__global__ void copy_f32_kernel(float* __restrict__ dst,
                                const float* __restrict__ src, size_t n4) {
    const float4* s = (const float4*)src;
    float4* d = (float4*)dst;
    for (size_t i = (size_t)blockIdx.x * blockDim.x + threadIdx.x; i < n4;
         i += (size_t)gridDim.x * blockDim.x) {
        d[i] = s[i];
    }
}

// ---------------------------------------------------------------------------
// Edge scatter-add: agg[dst[e]][:] += h[src[e]][:]   (agg pre-seeded with h)
// ---------------------------------------------------------------------------
__global__ void scatter_add_kernel(const float* __restrict__ h,
                                   const long long* __restrict__ src,
                                   const long long* __restrict__ dst,
                                   float* __restrict__ agg,
                                   long long E, int F) {
    const int chunks = F >> 2;
    const size_t total = (size_t)E * (size_t)chunks;
    for (size_t i = (size_t)blockIdx.x * blockDim.x + threadIdx.x; i < total;
         i += (size_t)gridDim.x * blockDim.x) {
        const size_t e = i / (size_t)chunks;
        const int    c = (int)(i - e * (size_t)chunks);
        const long long s = src[e];
        const long long d = dst[e];
        const float4 v = *reinterpret_cast<const float4*>(&h[(size_t)s * F + c * 4]);
        float* p = &agg[(size_t)d * F + c * 4];
        __hip_atomic_fetch_add(p + 0, v.x, __ATOMIC_RELAXED, __HIP_MEMORY_SCOPE_AGENT);
        __hip_atomic_fetch_add(p + 1, v.y, __ATOMIC_RELAXED, __HIP_MEMORY_SCOPE_AGENT);
        __hip_atomic_fetch_add(p + 2, v.z, __ATOMIC_RELAXED, __HIP_MEMORY_SCOPE_AGENT);
        __hip_atomic_fetch_add(p + 3, v.w, __ATOMIC_RELAXED, __HIP_MEMORY_SCOPE_AGENT);
    }
}

// ---------------------------------------------------------------------------
// TDM: DMA a [64 rows x 32 cols] bf16 tile of Bt[Nout][K] into LDS.
// 2D descriptor: data_size=2B, tile_dim0=32 (k), tile_dim1=64 (n),
// tensor_dim0_stride=K. Rows land back-to-back in LDS = Bs[64][32] layout.
// ---------------------------------------------------------------------------
#if defined(HAVE_TDM)
__device__ inline void tdm_load_b_tile(const unsigned short* tile_ptr,
                                       unsigned lds_off, int K, int Nout) {
    const unsigned long long ga = (unsigned long long)tile_ptr;
    v4u g0;
    g0.x = 1u;                                        // count=1, user desc
    g0.y = lds_off;                                   // lds_addr (bytes)
    g0.z = (unsigned)(ga & 0xFFFFFFFFu);              // global_addr[31:0]
    g0.w = (unsigned)((ga >> 32) & 0x01FFFFFFu)       // global_addr[56:32]
         | (2u << 30);                                // type=2 ("image")
    v8i_t g1;
    g1[0] = (int)(1u << 16);                          // data_size=1 -> 2 bytes
    g1[1] = (int)(((unsigned)K & 0xFFFFu) << 16);     // tensor_dim0[15:0]
    g1[2] = (int)((((unsigned)K >> 16) & 0xFFFFu)     // tensor_dim0[31:16]
         | (((unsigned)Nout & 0xFFFFu) << 16));       // tensor_dim1[15:0]
    g1[3] = (int)((((unsigned)Nout >> 16) & 0xFFFFu)  // tensor_dim1[31:16]
         | (32u << 16));                              // tile_dim0 = 32
    g1[4] = 64;                                       // tile_dim1=64, tile_dim2=0
    g1[5] = K;                                        // tensor_dim0_stride[31:0]
    g1[6] = 0;                                        // stride hi / dim1_stride lo
    g1[7] = 0;
    v4i_t z4 = {0, 0, 0, 0};
#if defined(TDM_6ARG)
    v8i_t z8 = {0, 0, 0, 0, 0, 0, 0, 0};
    __builtin_amdgcn_tensor_load_to_lds(g0, g1, z4, z4, z8, 0);
#else
    __builtin_amdgcn_tensor_load_to_lds(g0, g1, z4, z4, 0);
#endif
}
#endif

__device__ inline void wait_tensorcnt0() {
#if defined(HAVE_TENSORCNT)
    __builtin_amdgcn_s_wait_tensorcnt((short)0);
#else
    asm volatile("s_wait_tensorcnt 0x0" ::: "memory");
#endif
}

// ---------------------------------------------------------------------------
// Tiled WMMA GEMM:  out[M,Nout] = act( A[M,K](f32->bf16) @ W + bias )
// W supplied as bf16, pre-transposed: Bt[Nout][K].
// Block: 256 threads (8 waves), tile 128x64, K-step 32.
// Each wave: 16 rows x 64 cols = 4 x v_wmma_f32_16x16x32_bf16.
// K multiple of 32, Nout multiple of 64 (here 128/512/64).
// ---------------------------------------------------------------------------
__global__ void __launch_bounds__(256)
wmma_gemm_bias_kernel(const float* __restrict__ A,
                      const unsigned short* __restrict__ Bt,  // bf16 [Nout][K]
                      const float* __restrict__ bias,
                      float* __restrict__ out,
                      int M, int K, int Nout, int do_relu) {
    __shared__ unsigned short As[128][32];   // bf16 bits, [m][k]
    __shared__ unsigned short Bs[64][32];    // bf16 bits, [n][k]

    const int tid  = threadIdx.x;
    const int w    = tid >> 5;        // wave id 0..7 -> 16-row sub-tile
    const int lane = tid & 31;
    const int half = lane >> 4;       // 0: lanes 0-15, 1: lanes 16-31
    const int mr   = lane & 15;

    const int row0  = blockIdx.y * 128;
    const int ncol0 = blockIdx.x * 64;

#if defined(HAVE_TDM)
    const unsigned lds_bs = (unsigned)(size_t)(&Bs[0][0]); // LDS byte offset
#endif

    v8f acc[4];
    #pragma unroll
    for (int ct = 0; ct < 4; ++ct) acc[ct] = (v8f)(0.0f);

    for (int k0 = 0; k0 < K; k0 += 32) {
        // ---- stage A tile: 128x32 f32 -> bf16, float4 loads, b64 LDS stores
        #pragma unroll
        for (int i = 0; i < 4; ++i) {
            const int id = tid + i * 256;           // 0..1023
            const int mm = id >> 3;                 // 0..127
            const int q  = id & 7;                  // float4 chunk (4 floats)
            int gm = row0 + mm;
            gm = gm < M ? gm : (M - 1);             // clamp: no exec masking
            const float4 v = *reinterpret_cast<const float4*>(
                &A[(size_t)gm * K + (k0 + q * 4)]);
            uint2 p;
            p.x = pack_bf16x2(v.x, v.y);
            p.y = pack_bf16x2(v.z, v.w);
            *reinterpret_cast<uint2*>(&As[mm][q * 4]) = p;
        }
        // ---- stage B tile: 64x32 bf16 via Tensor Data Mover (wave 0) ----
#if defined(HAVE_TDM)
        if (tid < 32) {
            tdm_load_b_tile(&Bt[(size_t)ncol0 * K + k0], lds_bs, K, Nout);
            wait_tensorcnt0();
        }
#else
        {   // fallback: 256 threads x uint4 (16B) = 4KB
            const int n = tid >> 2;
            const int c = tid & 3;
            const uint4 v = *reinterpret_cast<const uint4*>(
                &Bt[(size_t)(ncol0 + n) * K + k0 + c * 8]);
            *reinterpret_cast<uint4*>(&Bs[n][c * 8]) = v;
        }
#endif
        __syncthreads();

        // ---- A fragment (ISA 16-bit A 16x32 layout) ----
        // lanes 0-15: K{0..7,16..23}; lanes 16-31: K{8..15,24..31}
        union { v16bf v; unsigned u[8]; } fa;
        {
            const unsigned* Arow =
                reinterpret_cast<const unsigned*>(&As[w * 16 + mr][0]);
            const int kb = half * 4;
            #pragma unroll
            for (int j = 0; j < 4; ++j) {
                fa.u[j]     = Arow[kb + j];
                fa.u[4 + j] = Arow[kb + 8 + j];
            }
        }

        // ---- 4 column tiles ----
        #pragma unroll
        for (int ct = 0; ct < 4; ++ct) {
            // B fragment: lanes 0-15 K{0..7,16..23}, lanes 16-31 K{8..15,24..31}, col = mr
            union { v16bf v; unsigned u[8]; } fb;
            const unsigned* Brow =
                reinterpret_cast<const unsigned*>(&Bs[ct * 16 + mr][0]);
            const int kb = half * 4;
            #pragma unroll
            for (int j = 0; j < 4; ++j) {
                fb.u[j]     = Brow[kb + j];
                fb.u[4 + j] = Brow[kb + 8 + j];
            }

            acc[ct] = __builtin_amdgcn_wmma_f32_16x16x32_bf16(
                false, fa.v, false, fb.v, (short)0, acc[ct], false, false);
        }
        __syncthreads();
    }

    // ---- epilogue: bias + optional ReLU ----
    #pragma unroll
    for (int ct = 0; ct < 4; ++ct) {
        const int n = ncol0 + ct * 16 + mr;
        const float b = bias[n];
        #pragma unroll
        for (int r = 0; r < 8; ++r) {
            const int m = row0 + w * 16 + r + half * 8;
            if (m < M) {
                float v = acc[ct][r] + b;
                if (do_relu) v = v > 0.0f ? v : 0.0f;
                out[(size_t)m * Nout + n] = v;
            }
        }
    }
}

// ---------------------------------------------------------------------------
// Launch
// ---------------------------------------------------------------------------
extern "C" void kernel_launch(void* const* d_in, const int* in_sizes, int n_in,
                              void* d_out, int out_size, void* d_ws, size_t ws_size,
                              hipStream_t stream) {
    const int F_IN = 128, H = 512, C = 64;
    const int N = in_sizes[0] / F_IN;
    const long long E = in_sizes[1] / 2;

    const float*     x    = (const float*)d_in[0];
    const long long* ei   = (const long long*)d_in[1];
    const long long* esrc = ei;
    const long long* edst = ei + E;
    const float* W1a = (const float*)d_in[2];
    const float* b1a = (const float*)d_in[3];
    const float* W1b = (const float*)d_in[4];
    const float* b1b = (const float*)d_in[5];
    const float* W2a = (const float*)d_in[6];
    const float* b2a = (const float*)d_in[7];
    const float* W2b = (const float*)d_in[8];
    const float* b2b = (const float*)d_in[9];
    const float* Wc  = (const float*)d_in[10];
    const float* bc  = (const float*)d_in[11];
    float* out = (float*)d_out;

    // workspace layout
    float* bufA = (float*)d_ws;                       // z buffers
    float* bufB = bufA + (size_t)N * H;               // intermediate t
    float* bufC = bufB + (size_t)N * H;               // layer outputs h
    unsigned short* w1a = (unsigned short*)(bufC + (size_t)N * H);
    unsigned short* w1b = w1a + (size_t)F_IN * H;
    unsigned short* w2a = w1b + (size_t)H * H;
    unsigned short* w2b = w2a + (size_t)H * H;
    unsigned short* wc  = w2b + (size_t)H * H;

    // --- weights -> bf16, transposed to [Nout][K] ---
    cvt_bf16_transpose_kernel<<<256, 256, 0, stream>>>(W1a, w1a, F_IN, H);
    cvt_bf16_transpose_kernel<<<1024, 256, 0, stream>>>(W1b, w1b, H, H);
    cvt_bf16_transpose_kernel<<<1024, 256, 0, stream>>>(W2a, w2a, H, H);
    cvt_bf16_transpose_kernel<<<1024, 256, 0, stream>>>(W2b, w2b, H, H);
    cvt_bf16_transpose_kernel<<<128, 256, 0, stream>>>(Wc, wc, H, C);

    const int scatter_blocks = 65536;

    // --- layer 1: z1 = x + segment_sum(x[src] -> dst) ---
    copy_f32_kernel<<<2048, 256, 0, stream>>>(bufA, x, (size_t)N * F_IN / 4);
    scatter_add_kernel<<<scatter_blocks, 256, 0, stream>>>(x, esrc, edst, bufA, E, F_IN);
    {   // t1 = relu(z1 @ W1a + b1a)
        dim3 g((H + 63) / 64, (N + 127) / 128);
        wmma_gemm_bias_kernel<<<g, 256, 0, stream>>>(bufA, w1a, b1a, bufB, N, F_IN, H, 1);
    }
    {   // h1 = relu(t1 @ W1b + b1b)   (GIN output + outer ReLU fused)
        dim3 g((H + 63) / 64, (N + 127) / 128);
        wmma_gemm_bias_kernel<<<g, 256, 0, stream>>>(bufB, w1b, b1b, bufC, N, H, H, 1);
    }

    // --- layer 2: z2 = h1 + segment_sum(h1[src] -> dst) ---
    copy_f32_kernel<<<4096, 256, 0, stream>>>(bufA, bufC, (size_t)N * H / 4);
    scatter_add_kernel<<<scatter_blocks, 256, 0, stream>>>(bufC, esrc, edst, bufA, E, H);
    {   // t2 = relu(z2 @ W2a + b2a)
        dim3 g((H + 63) / 64, (N + 127) / 128);
        wmma_gemm_bias_kernel<<<g, 256, 0, stream>>>(bufA, w2a, b2a, bufB, N, H, H, 1);
    }
    {   // h2 = relu(t2 @ W2b + b2b)
        dim3 g((H + 63) / 64, (N + 127) / 128);
        wmma_gemm_bias_kernel<<<g, 256, 0, stream>>>(bufB, w2b, b2b, bufC, N, H, H, 1);
    }

    // --- classifier: logits = h2 @ Wc + bc ---
    {
        dim3 g((C + 63) / 64, (N + 127) / 128);
        wmma_gemm_bias_kernel<<<g, 256, 0, stream>>>(bufC, wc, bc, out, N, H, C, 0);
    }
}
